// AGCRNCell_v8_29472065585949
// MI455X (gfx1250) — matrix operand order repaired and checked by reading, not compiled, not run on hardware
//
#include <hip/hip_runtime.h>

// ---------------------------------------------------------------------------
// AGCRN cell for MI455X (gfx1250): bf16 WMMA pipeline, f32 accumulation.
// N=2048 nodes, B=16 batch, C=128 concat channels, D=10 embed, CHEB_K=3.
// Heavy GEMMs (2048^3) stage tiles into double-buffered LDS with
// GLOBAL_LOAD_ASYNC_TO_LDS_B128 directly in WMMA operand layout; per-node
// contractions keep A operands register-resident and stream weights from L2.
// ---------------------------------------------------------------------------

typedef __attribute__((ext_vector_type(16))) __bf16        v16bf;
typedef __attribute__((ext_vector_type(8)))  float         v8f;
typedef __attribute__((ext_vector_type(4)))  unsigned int  u32x4;

#define NN 2048
#define DD 10

struct OpPair { u32x4 a, b; };

static __device__ __forceinline__ v16bf mkop(u32x4 a, u32x4 b) {
  OpPair p{a, b};
  return __builtin_bit_cast(v16bf, p);
}

static __device__ __forceinline__ unsigned short f2bf(float f) {
  unsigned int u = __builtin_bit_cast(unsigned int, f);
  u += 0x7FFFu + ((u >> 16) & 1u);          // round-to-nearest-even
  return (unsigned short)(u >> 16);
}

static __device__ __forceinline__ v8f zero8() {
  v8f z = {0.f, 0.f, 0.f, 0.f, 0.f, 0.f, 0.f, 0.f};
  return z;
}

static __device__ __forceinline__ v8f wmma_bf16(v16bf a, v16bf b, v8f c) {
  return __builtin_amdgcn_wmma_f32_16x16x32_bf16(false, a, false, b, (short)0, c,
                                                 false, false);
}

// Async global->LDS copy, 16 bytes per lane, per-lane LDS destination.
// Tracked with ASYNCcnt (ISA 10.7 / 15.18.3).
static __device__ __forceinline__ void async_b128(unsigned lds_off,
                                                  const void* gptr) {
  asm volatile("global_load_async_to_lds_b128 %0, %1, off"
               :
               : "v"(lds_off), "v"((unsigned long long)gptr)
               : "memory");
}

static __device__ __forceinline__ void wait_async4() {
  asm volatile("s_wait_asynccnt 0x4" ::: "memory");  // previous batch retired
}

static __device__ __forceinline__ unsigned lds_off32(const void* p) {
  return (unsigned)(unsigned long long)p;   // low 32 bits = LDS byte offset
}

// ---------------------------------------------------------------------------
// supports = softmax(relu(E @ E^T)) per row, stored bf16 [n][m]
// ---------------------------------------------------------------------------
__global__ __launch_bounds__(256) void k_supports(const float* __restrict__ E,
                                                  unsigned short* __restrict__ S1) {
  __shared__ float e[DD];
  __shared__ float row[NN];
  __shared__ float red[256];
  int n = blockIdx.x, tid = threadIdx.x;
  if (tid < DD) e[tid] = E[n * DD + tid];
  __syncthreads();
  float lmax = 0.0f;
  for (int m = tid; m < NN; m += 256) {
    float s = 0.f;
#pragma unroll
    for (int d = 0; d < DD; ++d) s += e[d] * E[m * DD + d];
    s = fmaxf(s, 0.f);
    row[m] = s;
    lmax = fmaxf(lmax, s);
  }
  red[tid] = lmax; __syncthreads();
  for (int off = 128; off > 0; off >>= 1) {
    if (tid < off) red[tid] = fmaxf(red[tid], red[tid + off]);
    __syncthreads();
  }
  float mx = red[0]; __syncthreads();
  float lsum = 0.f;
  for (int m = tid; m < NN; m += 256) {
    float v = __expf(row[m] - mx);
    row[m] = v;
    lsum += v;
  }
  red[tid] = lsum; __syncthreads();
  for (int off = 128; off > 0; off >>= 1) {
    if (tid < off) red[tid] += red[tid + off];
    __syncthreads();
  }
  float inv = 1.0f / red[0];
  for (int m = tid; m < NN; m += 256) S1[(size_t)n * NN + m] = f2bf(row[m] * inv);
}

// ---------------------------------------------------------------------------
// bf16 2048x2048 transpose (LDS tiled)
// ---------------------------------------------------------------------------
__global__ __launch_bounds__(256) void k_transpose(const unsigned short* __restrict__ in,
                                                   unsigned short* __restrict__ out) {
  __shared__ unsigned short tile[64][65];
  int tx = threadIdx.x & 63, ty = threadIdx.x >> 6;
  int bx = blockIdx.x * 64, by = blockIdx.y * 64;
  for (int r = ty; r < 64; r += 4)
    tile[r][tx] = in[(size_t)(by + r) * NN + bx + tx];
  __syncthreads();
  for (int r = ty; r < 64; r += 4)
    out[(size_t)(bx + r) * NN + by + tx] = tile[tx][r];
}

// ---------------------------------------------------------------------------
// Pack X1r[m][b*128+c] = bf16(concat(x,state))
// ---------------------------------------------------------------------------
__global__ __launch_bounds__(256) void k_pack_x1(const float* __restrict__ x,
                                                 const float* __restrict__ st,
                                                 unsigned short* __restrict__ X1r) {
  int idx = blockIdx.x * 256 + threadIdx.x;   // NN*2048
  int m = idx >> 11, bc = idx & 2047;
  int b = bc >> 7, c = bc & 127;
  float v = (c < 64) ? x[((size_t)b * NN + m) * 64 + c]
                     : st[((size_t)b * NN + m) * 64 + (c - 64)];
  X1r[idx] = f2bf(v);
}

// x-part of candidate: X2r[m][b*128+c] = bf16(x), c<64 (z_r part comes later)
__global__ __launch_bounds__(256) void k_pack_x2(const float* __restrict__ x,
                                                 unsigned short* __restrict__ X2r) {
  int idx = blockIdx.x * 256 + threadIdx.x;   // NN*16*64
  int m = idx >> 10; int rem = idx & 1023; int b = rem >> 6; int c = rem & 63;
  X2r[(size_t)m * 2048 + b * 128 + c] = f2bf(x[((size_t)b * NN + m) * 64 + c]);
}

// wpool [10][3][128][O] f32 -> WT [d][O][384] bf16 (B-operand friendly)
__global__ void k_prep_wt(const float* __restrict__ wp,
                          unsigned short* __restrict__ WT, int O, int total) {
  int idx = blockIdx.x * 256 + threadIdx.x;
  if (idx >= total) return;
  int d = idx / (384 * O); int rem = idx - d * 384 * O;
  int kc = rem / O; int o = rem - kc * O;
  WT[((size_t)d * O + o) * 384 + kc] = f2bf(wp[((size_t)d * 384 + kc) * O + o]);
}

// Bias[n][o] = sum_d E[n,d] * bpool[d,o]
__global__ void k_prep_bias(const float* __restrict__ E, const float* __restrict__ bp,
                            float* __restrict__ Bias, int O) {
  int idx = blockIdx.x * 256 + threadIdx.x;   // NN*O
  int n = idx / O, o = idx - n * O;
  float s = 0.f;
#pragma unroll
  for (int d = 0; d < DD; ++d) s += E[n * DD + d] * bp[d * O + o];
  Bias[idx] = s;
}

__global__ void k_prep_lw(const float* __restrict__ lw, unsigned short* __restrict__ LW) {
  int idx = blockIdx.x * 256 + threadIdx.x;   // 64*128
  LW[idx] = f2bf(lw[idx]);
}

// ---------------------------------------------------------------------------
// C = op(A @ B)  : A[2048x2048] row-major bf16, BT[2048x2048] = B^T row-major,
// C bf16.  mode 1: C = 2*acc - I (Chebyshev).  Block tile 128x128, K-step 32.
// Double-buffered LDS; tiles land already in WMMA operand order (ISA 7.12.2)
// via GLOBAL_LOAD_ASYNC_TO_LDS_B128 with per-lane LDS destinations. The
// in-order ASYNCcnt lets us wait with count 4 so the previous buffer's loads
// are fully hidden behind the current buffer's WMMAs.
// ---------------------------------------------------------------------------
__global__ __launch_bounds__(256) void k_gemm(const unsigned short* __restrict__ A,
                                              const unsigned short* __restrict__ BT,
                                              unsigned short* __restrict__ C, int mode) {
  __shared__ __align__(16) unsigned short lA[2 * 4096];   // 2 x 128x32 bf16
  __shared__ __align__(16) unsigned short lB[2 * 4096];
  int tid = threadIdx.x, lane = tid & 31, wave = tid >> 5;
  int m0 = blockIdx.y * 128, n0 = blockIdx.x * 128;

  // Per-thread staging chunks: 512 16-byte chunks per operand tile, 2 each.
  const unsigned short* pA[2];
  const unsigned short* pB[2];
  unsigned ldsA[2], ldsB[2];
#pragma unroll
  for (int i = 0; i < 2; ++i) {
    int cid = tid + i * 256;
    int row = cid >> 2, kk8 = cid & 3;
    int mT = row >> 4, r = row & 15;
    // A operand: lane<16 -> K {0..7,16..23}; lane>=16 -> {8..15,24..31}
    int laneA = r + ((kk8 & 1) ? 16 : 0);
    int hA = (kk8 < 2) ? 0 : 8;
    ldsA[i] = lds_off32(lA + ((mT * 32 + laneA) * 16 + hA));
    pA[i] = A + (size_t)(m0 + row) * NN + kk8 * 8;
    // B operand: lane<16 -> K {0..7,16..23} (col=lane); lane>=16 -> {8..15,24..31}
    int laneB = r + ((kk8 & 1) ? 16 : 0);
    int hB = (kk8 < 2) ? 0 : 8;
    ldsB[i] = lds_off32(lB + ((mT * 32 + laneB) * 16 + hB));
    pB[i] = BT + (size_t)(n0 + row) * NN + kk8 * 8;
  }

  auto issue = [&](int buf) {
    unsigned boff = (unsigned)buf * 8192;   // 4096 halves per buffer
#pragma unroll
    for (int i = 0; i < 2; ++i) {
      async_b128(ldsA[i] + boff, pA[i]);
      async_b128(ldsB[i] + boff, pB[i]);
      __builtin_prefetch(pA[i] + 32, 0, 1);
      __builtin_prefetch(pB[i] + 32, 0, 1);
      pA[i] += 32;
      pB[i] += 32;
    }
  };

  v8f acc[8];
#pragma unroll
  for (int t = 0; t < 8; ++t) acc[t] = zero8();

  issue(0);
  int cur = 0;
  for (int it = 0; it < 64; ++it) {
    issue(cur ^ 1);           // next tile (65th batch is dead but harmless)
    wait_async4();            // own previous batch has retired (in-order)
    __syncthreads();          // everyone's current buffer is visible

    const unsigned short* ap = lA + cur * 4096 + (wave * 32 + lane) * 16;
    v16bf av = mkop(*(const u32x4*)ap, *(const u32x4*)(ap + 8));
    v16bf bv[8];
#pragma unroll
    for (int t = 0; t < 8; ++t) {
      const unsigned short* bp = lB + cur * 4096 + (t * 32 + lane) * 16;
      bv[t] = mkop(*(const u32x4*)bp, *(const u32x4*)(bp + 8));
    }
#pragma unroll
    for (int t = 0; t < 8; ++t) acc[t] = wmma_bf16(av, bv[t], acc[t]);
    __syncthreads();          // done reading cur before it is overwritten
    cur ^= 1;
  }

  int r15 = lane & 15, hi8 = (lane >> 4) * 8;
#pragma unroll
  for (int t = 0; t < 8; ++t) {
#pragma unroll
    for (int v = 0; v < 8; ++v) {
      int row = m0 + wave * 16 + v + hi8;
      int col = n0 + t * 16 + r15;
      float xv = acc[t][v];
      if (mode == 1) xv = 2.0f * xv - (row == col ? 1.0f : 0.0f);
      C[(size_t)row * NN + col] = f2bf(xv);
    }
  }
}

// ---------------------------------------------------------------------------
// Gate combine: per node n (one wave each):
//   g[b,o] = sigmoid( sum_d E[n,d] * (A_n @ Wd) + bias )   A_n: 16x384
//   z[b,j] = g @ lin_w^T + lin_b  ->  X2r[n][b*128 + 64 + j]
// A operands (12 x v16bf) are preloaded once and stay register-resident.
// ---------------------------------------------------------------------------
__global__ __launch_bounds__(256) void k_combine_gate(
    const unsigned short* __restrict__ S0r,  // X1r  [n][bc]  (k=0 slab)
    const unsigned short* __restrict__ XG1,  // S1@X [n][bc]
    const unsigned short* __restrict__ XG2,  // S2@X [n][bc]
    const unsigned short* __restrict__ WT,   // [d][128][384] bf16
    const float* __restrict__ Bias,          // [n][128]
    const float* __restrict__ E,             // [n][10]
    const unsigned short* __restrict__ LW,   // [64][128] bf16 (= lin_w)
    const float* __restrict__ linb,          // [64]
    unsigned short* __restrict__ X2r) {
  __shared__ __align__(16) unsigned short sig[8 * 16 * 128];  // per-wave 16x128
  int tid = threadIdx.x, lane = tid & 31, wave = tid >> 5;
  int n = blockIdx.x * 8 + wave;
  int r = lane & 15;
  int hA = (lane < 16) ? 0 : 16;  // A operand K-half select
  int kB = (lane < 16) ? 0 : 16;  // B operand K base
  const unsigned short* slab[3] = {S0r + (size_t)n * 2048, XG1 + (size_t)n * 2048,
                                   XG2 + (size_t)n * 2048};
  v16bf aop[12];
#pragma unroll
  for (int ks = 0; ks < 12; ++ks) {          // K = 3*128 = 384
    int kc0 = ks * 32;
    const unsigned short* ap = slab[kc0 >> 7] + r * 128 + (kc0 & 127) + hA;
    aop[ks] = mkop(*(const u32x4*)ap, *(const u32x4*)(ap + 8));
  }

  for (int t = 0; t < 8; ++t) {
    float bias = Bias[n * 128 + t * 16 + r];
    v8f acc = zero8();
    for (int d = 0; d < DD; ++d) {
      float ed = E[n * DD + d];
      v8f tmp0 = zero8(), tmp1 = zero8();
#pragma unroll
      for (int ks = 0; ks < 12; ks += 2) {
        const unsigned short* bp0 =
            WT + ((size_t)d * 128 + t * 16 + r) * 384 + ks * 32 + kB;
        v16bf bv0 = mkop(*(const u32x4*)bp0, *(const u32x4*)(bp0 + 8));
        const unsigned short* bp1 = bp0 + 32;
        v16bf bv1 = mkop(*(const u32x4*)bp1, *(const u32x4*)(bp1 + 8));
        tmp0 = wmma_bf16(aop[ks], bv0, tmp0);
        tmp1 = wmma_bf16(aop[ks + 1], bv1, tmp1);
      }
      acc += (tmp0 + tmp1) * ed;
    }
#pragma unroll
    for (int v = 0; v < 8; ++v) {
      float sg = 1.0f / (1.0f + __expf(-(acc[v] + bias)));
      int b = v + ((lane < 16) ? 0 : 8);
      sig[wave * 2048 + b * 128 + t * 16 + r] = f2bf(sg);
    }
  }
  __syncthreads();

  // z = sig(16x128) @ lin_w^T + lin_b   (K=128, N=64)
  for (int tj = 0; tj < 4; ++tj) {
    v8f z0 = zero8(), z1 = zero8();
#pragma unroll
    for (int ks = 0; ks < 4; ks += 2) {
      const unsigned short* ap0 = sig + wave * 2048 + r * 128 + ks * 32 + hA;
      v16bf av0 = mkop(*(const u32x4*)ap0, *(const u32x4*)(ap0 + 8));
      const unsigned short* ap1 = ap0 + 32;
      v16bf av1 = mkop(*(const u32x4*)ap1, *(const u32x4*)(ap1 + 8));
      const unsigned short* bp0 = LW + (tj * 16 + r) * 128 + ks * 32 + kB;
      v16bf bv0 = mkop(*(const u32x4*)bp0, *(const u32x4*)(bp0 + 8));
      const unsigned short* bp1 = bp0 + 32;
      v16bf bv1 = mkop(*(const u32x4*)bp1, *(const u32x4*)(bp1 + 8));
      z0 = wmma_bf16(av0, bv0, z0);
      z1 = wmma_bf16(av1, bv1, z1);
    }
    v8f z = z0 + z1;
    float lb = linb[tj * 16 + r];
#pragma unroll
    for (int v = 0; v < 8; ++v) {
      int b = v + ((lane < 16) ? 0 : 8);
      X2r[(size_t)n * 2048 + b * 128 + 64 + tj * 16 + r] = f2bf(z[v] + lb);
    }
  }
}

// ---------------------------------------------------------------------------
// Update combine: hc[b,n,o] = tanh( sum_d E[n,d]*(A_n @ Wd) + bias ), O=64
// ---------------------------------------------------------------------------
__global__ __launch_bounds__(256) void k_combine_upd(
    const unsigned short* __restrict__ S0r,  // X2r
    const unsigned short* __restrict__ XG1,  // S1@X2
    const unsigned short* __restrict__ XG2,  // S2@X2
    const unsigned short* __restrict__ WT,   // [d][64][384] bf16
    const float* __restrict__ Bias,          // [n][64]
    const float* __restrict__ E,
    float* __restrict__ out) {               // [B][N][64] f32
  int tid = threadIdx.x, lane = tid & 31, wave = tid >> 5;
  int n = blockIdx.x * 8 + wave;
  int r = lane & 15;
  int hA = (lane < 16) ? 0 : 16;
  int kB = (lane < 16) ? 0 : 16;
  const unsigned short* slab[3] = {S0r + (size_t)n * 2048, XG1 + (size_t)n * 2048,
                                   XG2 + (size_t)n * 2048};
  v16bf aop[12];
#pragma unroll
  for (int ks = 0; ks < 12; ++ks) {
    int kc0 = ks * 32;
    const unsigned short* ap = slab[kc0 >> 7] + r * 128 + (kc0 & 127) + hA;
    aop[ks] = mkop(*(const u32x4*)ap, *(const u32x4*)(ap + 8));
  }

  for (int t = 0; t < 4; ++t) {
    float bias = Bias[n * 64 + t * 16 + r];
    v8f acc = zero8();
    for (int d = 0; d < DD; ++d) {
      float ed = E[n * DD + d];
      v8f tmp0 = zero8(), tmp1 = zero8();
#pragma unroll
      for (int ks = 0; ks < 12; ks += 2) {
        const unsigned short* bp0 =
            WT + ((size_t)d * 64 + t * 16 + r) * 384 + ks * 32 + kB;
        v16bf bv0 = mkop(*(const u32x4*)bp0, *(const u32x4*)(bp0 + 8));
        const unsigned short* bp1 = bp0 + 32;
        v16bf bv1 = mkop(*(const u32x4*)bp1, *(const u32x4*)(bp1 + 8));
        tmp0 = wmma_bf16(aop[ks], bv0, tmp0);
        tmp1 = wmma_bf16(aop[ks + 1], bv1, tmp1);
      }
      acc += (tmp0 + tmp1) * ed;
    }
#pragma unroll
    for (int v = 0; v < 8; ++v) {
      int b = v + ((lane < 16) ? 0 : 8);
      out[(size_t)b * NN * 64 + (size_t)n * 64 + t * 16 + r] = tanhf(acc[v] + bias);
    }
  }
}

// ---------------------------------------------------------------------------
extern "C" void kernel_launch(void* const* d_in, const int* in_sizes, int n_in,
                              void* d_out, int out_size, void* d_ws, size_t ws_size,
                              hipStream_t stream) {
  const float* x   = (const float*)d_in[0];
  const float* st  = (const float*)d_in[1];
  const float* E   = (const float*)d_in[2];
  const float* gwp = (const float*)d_in[3];
  const float* gbp = (const float*)d_in[4];
  const float* uwp = (const float*)d_in[5];
  const float* ubp = (const float*)d_in[6];
  const float* lw  = (const float*)d_in[7];
  const float* lb  = (const float*)d_in[8];

  char* ws = (char*)d_ws;
  const size_t SZ = (size_t)NN * NN * 2;   // one bf16 2048x2048 matrix = 8 MB
  unsigned short* S1   = (unsigned short*)(ws + 0 * SZ);
  unsigned short* S1T  = (unsigned short*)(ws + 1 * SZ);
  unsigned short* S2   = (unsigned short*)(ws + 2 * SZ);
  unsigned short* X1r  = (unsigned short*)(ws + 3 * SZ);
  unsigned short* X1T  = (unsigned short*)(ws + 4 * SZ);
  unsigned short* XG11 = (unsigned short*)(ws + 5 * SZ);
  unsigned short* XG12 = (unsigned short*)(ws + 6 * SZ);
  unsigned short* X2r  = (unsigned short*)(ws + 7 * SZ);
  unsigned short* X2T  = (unsigned short*)(ws + 8 * SZ);
  unsigned short* XG21 = (unsigned short*)(ws + 9 * SZ);
  unsigned short* XG22 = (unsigned short*)(ws + 10 * SZ);
  char* extra = ws + 11 * SZ;
  unsigned short* WT1 = (unsigned short*)(extra);               // 983040 B
  unsigned short* WT2 = (unsigned short*)(extra + 0x100000);    // 491520 B
  float* B1 = (float*)(extra + 0x200000);                       // 1 MB
  float* B2 = (float*)(extra + 0x300000);                       // 512 KB
  unsigned short* LW = (unsigned short*)(extra + 0x380000);     // 16 KB

  dim3 g16(16, 16, 1);
  dim3 g32(32, 32, 1);

  k_supports<<<NN, 256, 0, stream>>>(E, S1);
  k_transpose<<<g32, 256, 0, stream>>>(S1, S1T);
  k_pack_x1<<<(NN * 2048) / 256, 256, 0, stream>>>(x, st, X1r);
  k_transpose<<<g32, 256, 0, stream>>>(X1r, X1T);

  k_gemm<<<g16, 256, 0, stream>>>(S1, S1T, S2, 1);   // S2 = 2*S1@S1 - I
  k_gemm<<<g16, 256, 0, stream>>>(S1, X1T, XG11, 0); // S1 @ X1
  k_gemm<<<g16, 256, 0, stream>>>(S2, X1T, XG12, 0); // S2 @ X1

  k_prep_wt<<<(10 * 384 * 128 + 255) / 256, 256, 0, stream>>>(gwp, WT1, 128, 10 * 384 * 128);
  k_prep_wt<<<(10 * 384 * 64 + 255) / 256, 256, 0, stream>>>(uwp, WT2, 64, 10 * 384 * 64);
  k_prep_bias<<<(NN * 128) / 256, 256, 0, stream>>>(E, gbp, B1, 128);
  k_prep_bias<<<(NN * 64) / 256, 256, 0, stream>>>(E, ubp, B2, 64);
  k_prep_lw<<<(64 * 128) / 256, 256, 0, stream>>>(lw, LW);
  k_pack_x2<<<(NN * 16 * 64) / 256, 256, 0, stream>>>(x, X2r);

  k_combine_gate<<<NN / 8, 256, 0, stream>>>(X1r, XG11, XG12, WT1, B1, E, LW, lb, X2r);

  k_transpose<<<g32, 256, 0, stream>>>(X2r, X2T);
  k_gemm<<<g16, 256, 0, stream>>>(S1, X2T, XG21, 0); // S1 @ X2
  k_gemm<<<g16, 256, 0, stream>>>(S2, X2T, XG22, 0); // S2 @ X2

  k_combine_upd<<<NN / 8, 256, 0, stream>>>(X2r, XG21, XG22, WT2, B2, E, (float*)d_out);

  (void)in_sizes; (void)n_in; (void)out_size; (void)ws_size;
}
